// TransformerXL_1941325218040
// MI455X (gfx1250) — matrix operand order repaired and checked by reading, not compiled, not run on hardware
//
#include <hip/hip_runtime.h>
#include <math.h>

// ---------------- Model dims (Transformer-XL reference) ----------------
#define QLEN   512
#define BSZ    4
#define DMODEL 1024
#define NLAYER 4
#define NHEAD  16
#define DHEAD  64
#define DINNER 4096
#define MEMLEN 512
#define KLEN   1024      // MEMLEN + QLEN
#define HDH    1024      // NHEAD * DHEAD
#define QKV3   3072      // 3 * HDH

typedef unsigned short u16;

// ---------------- WMMA types (gfx1250, wave32) ----------------
typedef __attribute__((ext_vector_type(16))) __bf16 v16bf;
typedef __attribute__((ext_vector_type(8)))  float  v8f;

union BFrag { v16bf v; u16 s[16]; uint4 q[2]; };
union FAcc  { v8f   v; float f[8]; };

__device__ __forceinline__ u16 f2bf(float x) {
  union { float f; unsigned u; } c; c.f = x;
  unsigned r = c.u + 0x7FFFu + ((c.u >> 16) & 1u);   // round-to-nearest-even
  return (u16)(r >> 16);
}
__device__ __forceinline__ float bf2f(u16 s) {
  union { unsigned u; float f; } c; c.u = ((unsigned)s) << 16; return c.f;
}

// =======================================================================
// Straight convert f32 -> bf16, 8 elements/thread, b128 in / b128 out.
// =======================================================================
__global__ __launch_bounds__(256) void conv_bf16_kernel(
    const float* __restrict__ in, u16* __restrict__ out, int n8)
{
  const int i = blockIdx.x * blockDim.x + threadIdx.x;
  if (i >= n8) return;
  const float4* p = (const float4*)in + (size_t)i * 2;
  const float4 a = p[0], b = p[1];
  union { u16 s[8]; uint4 q; } r;
  r.s[0] = f2bf(a.x); r.s[1] = f2bf(a.y); r.s[2] = f2bf(a.z); r.s[3] = f2bf(a.w);
  r.s[4] = f2bf(b.x); r.s[5] = f2bf(b.y); r.s[6] = f2bf(b.z); r.s[7] = f2bf(b.w);
  ((uint4*)out)[i] = r.q;
}

// =======================================================================
// Weight transpose-convert: W[K][N] f32 -> WT[N][K] bf16 (LDS 32x32 tiles).
// =======================================================================
__global__ __launch_bounds__(256) void convtr_kernel(
    const float* __restrict__ in, u16* __restrict__ out, int K, int N)
{
  __shared__ float t[32][33];
  const int n0 = blockIdx.x * 32, k0 = blockIdx.y * 32;
  const int tx = threadIdx.x, ty = threadIdx.y;   // (32, 8)
#pragma unroll
  for (int r = 0; r < 32; r += 8)
    t[ty + r][tx] = in[(size_t)(k0 + ty + r) * N + n0 + tx];
  __syncthreads();
#pragma unroll
  for (int r = 0; r < 32; r += 8)
    out[(size_t)(n0 + ty + r) * K + k0 + tx] = f2bf(t[tx][ty + r]);
}

// =======================================================================
// V transpose (bf16 relayout): heads_bf v-part -> vT[b][h][d][key]
// =======================================================================
__global__ __launch_bounds__(256) void vtrans_kernel(
    const u16* __restrict__ heads_bf, u16* __restrict__ vT)
{
  __shared__ u16 t[64][65];
  const int key0 = blockIdx.x * 64;
  const int h = blockIdx.y, b = blockIdx.z;
  const int tid = threadIdx.x;
#pragma unroll
  for (int e = 0; e < 16; ++e) {
    const int idx = tid + e * 256;          // coalesced over d
    const int key = idx >> 6, d = idx & 63;
    t[key][d] = heads_bf[((size_t)(key0 + key) * BSZ + b) * QKV3 + 2 * HDH + h * DHEAD + d];
  }
  __syncthreads();
#pragma unroll
  for (int e = 0; e < 16; ++e) {
    const int idx = tid + e * 256;          // coalesced over key
    const int d = idx >> 6, key = idx & 63;
    vT[((size_t)(b * NHEAD + h) * DHEAD + d) * KLEN + key0 + key] = t[key][d];
  }
}

// =======================================================================
// Pure-bf16 WMMA GEMM: C[M,N] = op(A[M,K]bf16 @ WT[N,K]bf16^T + bias)
// One wave: 32(M) x 64(N) tile (8 accumulators) -> each B-fragment reused
// by 2 A-fragments (1.5 b128 loads per WMMA). K pipelined by 2 (ping-pong
// fragment buffers) so VMEM overlaps the matrix pipe (partial loadcnt
// waits instead of full drains). Unguarded speculative prefetch.
// =======================================================================
__global__ __launch_bounds__(32) void gemm_bf16_kernel(
    const u16* __restrict__ A0, int a0rows,
    const u16* __restrict__ A1,
    const u16* __restrict__ WT,
    const float* __restrict__ bias,
    float* __restrict__ Cf, u16* __restrict__ Cb,
    int N, int K, int doRelu)
{
  const int lane = threadIdx.x;
  const int lh   = lane & 15;
  const int hi   = lane & 16;
  const int j0   = blockIdx.x * 64;
  const int i0   = blockIdx.y * 32;

  FAcc acc[2][4];
#pragma unroll
  for (int m = 0; m < 2; ++m)
#pragma unroll
    for (int t = 0; t < 4; ++t)
#pragma unroll
      for (int g = 0; g < 8; ++g) acc[m][t].f[g] = 0.f;

  const u16* aptr[2];
#pragma unroll
  for (int m = 0; m < 2; ++m) {
    const int arow = i0 + m * 16 + lh;
    aptr[m] = ((arow < a0rows) ? (A0 + (size_t)arow * K)
                               : (A1 + (size_t)(arow - a0rows) * K)) + (hi ? 8 : 0);
  }
  const u16* bptr[4];
#pragma unroll
  for (int t = 0; t < 4; ++t)
    bptr[t] = WT + (size_t)(j0 + t * 16 + lh) * K + (hi ? 8 : 0);

  auto loadA = [&](BFrag fa[2], int k0) {
#pragma unroll
    for (int m = 0; m < 2; ++m) {
      fa[m].q[0] = *(const uint4*)(aptr[m] + k0);        // K {base..base+7}
      fa[m].q[1] = *(const uint4*)(aptr[m] + k0 + 16);   // K {base+16..+23}
    }
  };
  auto loadB = [&](BFrag fb[4], int k0) {
#pragma unroll
    for (int t = 0; t < 4; ++t) {
      fb[t].q[0] = *(const uint4*)(bptr[t] + k0);
      fb[t].q[1] = *(const uint4*)(bptr[t] + k0 + 16);
    }
  };

  BFrag fa0[2], fb0[4], fa1[2], fb1[4];
  loadA(fa0, 0);
  loadB(fb0, 0);

  for (int k0 = 0; k0 < K; k0 += 64) {
    loadA(fa1, k0 + 32);
    loadB(fb1, k0 + 32);
    __builtin_prefetch(bptr[0] + k0 + 64, 0, 1);   // speculative: safe if OOB
#pragma unroll
    for (int t = 0; t < 4; ++t) {
      acc[0][t].v = __builtin_amdgcn_wmma_f32_16x16x32_bf16(
          false, fa0[0].v, false, fb0[t].v, (short)0, acc[0][t].v, false, false);
      acc[1][t].v = __builtin_amdgcn_wmma_f32_16x16x32_bf16(
          false, fa0[1].v, false, fb0[t].v, (short)0, acc[1][t].v, false, false);
    }
    if (k0 + 64 < K) {
      loadA(fa0, k0 + 64);
      loadB(fb0, k0 + 64);
    }
#pragma unroll
    for (int t = 0; t < 4; ++t) {
      acc[0][t].v = __builtin_amdgcn_wmma_f32_16x16x32_bf16(
          false, fa1[0].v, false, fb1[t].v, (short)0, acc[0][t].v, false, false);
      acc[1][t].v = __builtin_amdgcn_wmma_f32_16x16x32_bf16(
          false, fa1[1].v, false, fb1[t].v, (short)0, acc[1][t].v, false, false);
    }
  }

#pragma unroll
  for (int m = 0; m < 2; ++m)
#pragma unroll
    for (int t = 0; t < 4; ++t)
#pragma unroll
      for (int g = 0; g < 8; ++g) {
        const int row = i0 + m * 16 + g + (hi ? 8 : 0);
        const int col = j0 + t * 16 + lh;
        float vv = acc[m][t].f[g];
        if (bias)   vv += bias[col];
        if (doRelu) vv = fmaxf(vv, 0.f);
        if (Cb) Cb[(size_t)row * N + col] = f2bf(vv);
        else    Cf[(size_t)row * N + col] = vv;
      }
}

// =======================================================================
// Fused Transformer-XL attention (flash-style; scores never hit memory).
// One wave per (16 query rows, head h, batch b).
//   S = (Q+u)K^T + rel_shift((Q+v)R^T); online softmax; O = P V.
// rel_shift[i,j] = BDpre[i, j+511-i]: computed as a 16x48 WMMA strip and
// extracted with ds_bpermute diagonal gathers. All score-side fragments
// are loaded in one batch (20 x b128) before the 10 WMMAs.
// =======================================================================
__global__ __launch_bounds__(32) void flash_attn_kernel(
    const u16* __restrict__ heads,  // [KLEN*BSZ][3072] bf16  (q|k|v)
    const u16* __restrict__ rk,     // [KLEN][HDH] bf16
    const u16* __restrict__ vT,     // [BSZ*NHEAD*DHEAD][KLEN] bf16
    const float* __restrict__ u,    // [NHEAD][DHEAD] f32
    const float* __restrict__ v,    // [NHEAD][DHEAD] f32
    u16* __restrict__ av)           // [QLEN*BSZ][HDH] bf16
{
  __shared__ u16 sP[16 * 40];       // P tile 16x32 bf16 (16B-aligned rows)

  const int lane = threadIdx.x;
  const int lh   = lane & 15;
  const int hi   = lane & 16;
  const int i0   = blockIdx.x * 16;
  const int h    = blockIdx.y;
  const int b    = blockIdx.z;

  // ---- Q fragments (+u, +v): done once, outside the K loop
  BFrag qu[2], qv[2];
  {
    const int qrow = i0 + lh;
    const u16* qp = heads + ((size_t)(MEMLEN + qrow) * BSZ + b) * QKV3 + h * DHEAD;
    const float* up = u + h * DHEAD;
    const float* vp = v + h * DHEAD;
    const int ab = hi ? 16 : 0;
#pragma unroll
    for (int s = 0; s < 2; ++s) {
#pragma unroll
      for (int c = 0; c < 2; ++c) {
        const int d0 = s * 32 + ab + c * 8;
#pragma unroll
        for (int e = 0; e < 8; ++e) {
          const float q = bf2f(qp[d0 + e]);
          qu[s].s[c * 8 + e] = f2bf(q + up[d0 + e]);
          qv[s].s[c * 8 + e] = f2bf(q + vp[d0 + e]);
        }
      }
    }
  }

  FAcc o[4];
#pragma unroll
  for (int t = 0; t < 4; ++t)
#pragma unroll
    for (int g = 0; g < 8; ++g) o[t].f[g] = 0.f;

  float mrow[8], lrow[8];
#pragma unroll
  for (int g = 0; g < 8; ++g) { mrow[g] = -INFINITY; lrow[g] = 0.f; }

  // mask: key j visible iff j - i <= MEMLEN  ->  stop at i0+528
  int jend = i0 + 15 + MEMLEN + 1;
  jend = (jend + 31) & ~31;
  if (jend > KLEN) jend = KLEN;

  const float scale = 0.125f;   // 1/sqrt(DHEAD)

  for (int j0 = 0; j0 < jend; j0 += 32) {
    // ---------- batch-load all score-side fragments (20 x b128) ----------
    BFrag kb[2][2];   // [key tile][k-step]
#pragma unroll
    for (int t = 0; t < 2; ++t) {
      const int key = j0 + t * 16 + lh;
      const u16* kp = heads + ((size_t)key * BSZ + b) * QKV3 + HDH + h * DHEAD + hi;
      kb[t][0].q[0] = *(const uint4*)(kp);
      kb[t][0].q[1] = *(const uint4*)(kp + 8);
      kb[t][1].q[0] = *(const uint4*)(kp + 32);
      kb[t][1].q[1] = *(const uint4*)(kp + 40);
    }
    const int r0 = j0 - i0 + 496;
    BFrag rb[3][2];   // [pre tile][k-step]
#pragma unroll
    for (int p = 0; p < 3; ++p) {
      int rr = r0 + p * 16 + lh;
      rr = rr < 0 ? 0 : (rr > KLEN - 1 ? KLEN - 1 : rr);  // OOB == masked
      const u16* rp = rk + (size_t)rr * HDH + h * DHEAD + hi;
      rb[p][0].q[0] = *(const uint4*)(rp);
      rb[p][0].q[1] = *(const uint4*)(rp + 8);
      rb[p][1].q[0] = *(const uint4*)(rp + 32);
      rb[p][1].q[1] = *(const uint4*)(rp + 40);
    }

    // ---------- AC = (Q+u) K^T : two 16-col tiles ----------
    FAcc s01[2];
#pragma unroll
    for (int t = 0; t < 2; ++t) {
#pragma unroll
      for (int g = 0; g < 8; ++g) s01[t].f[g] = 0.f;
#pragma unroll
      for (int s = 0; s < 2; ++s)
        s01[t].v = __builtin_amdgcn_wmma_f32_16x16x32_bf16(
            false, qu[s].v, false, kb[t][s].v, (short)0, s01[t].v, false, false);
    }

    // ---------- BD pre-shift strip: Ppre[16 x 48] = (Q+v) R^T ----------
    FAcc pre[3];
#pragma unroll
    for (int p = 0; p < 3; ++p) {
#pragma unroll
      for (int g = 0; g < 8; ++g) pre[p].f[g] = 0.f;
#pragma unroll
      for (int s = 0; s < 2; ++s)
        pre[p].v = __builtin_amdgcn_wmma_f32_16x16x32_bf16(
            false, qv[s].v, false, rb[p][s].v, (short)0, pre[p].v, false, false);
    }

    // ---------- diagonal extract + mask + online softmax ----------
#pragma unroll
    for (int g = 0; g < 8; ++g) {
      const int M   = g + (hi ? 8 : 0);
      const int cc  = lh + 15 - M;           // [0,30]
      const int idx = (cc & 15) | hi;
      const float e0a = __shfl(pre[0].f[g], idx, 32);
      const float e0b = __shfl(pre[1].f[g], idx, 32);
      const float e1b = __shfl(pre[2].f[g], idx, 32);
      const float bd0 = (cc < 16) ? e0a : e0b;
      const float bd1 = (cc < 16) ? e0b : e1b;

      const int row = i0 + M;
      const int jc0 = j0 + lh, jc1 = jc0 + 16;
      float sv0 = (jc0 - row > MEMLEN) ? -1e30f : (s01[0].f[g] + bd0) * scale;
      float sv1 = (jc1 - row > MEMLEN) ? -1e30f : (s01[1].f[g] + bd1) * scale;

      float tmax = fmaxf(sv0, sv1);          // row max across 16-lane half
      tmax = fmaxf(tmax, __shfl_xor(tmax, 1, 32));
      tmax = fmaxf(tmax, __shfl_xor(tmax, 2, 32));
      tmax = fmaxf(tmax, __shfl_xor(tmax, 4, 32));
      tmax = fmaxf(tmax, __shfl_xor(tmax, 8, 32));
      const float mnew  = fmaxf(mrow[g], tmax);
      const float alpha = __expf(mrow[g] - mnew);
      const float p0 = __expf(sv0 - mnew);
      const float p1 = __expf(sv1 - mnew);
      lrow[g] = lrow[g] * alpha + p0 + p1;   // per-lane partial row sum
      mrow[g] = mnew;
#pragma unroll
      for (int t = 0; t < 4; ++t) o[t].f[g] *= alpha;

      sP[M * 40 + lh]      = f2bf(p0);
      sP[M * 40 + 16 + lh] = f2bf(p1);
    }
    __syncthreads();

    // ---------- P (C layout) -> A fragment via LDS ----------
    BFrag fp;
    {
      const int kb2 = hi ? 16 : 0;
      fp.q[0] = *(const uint4*)(&sP[lh * 40 + kb2]);
      fp.q[1] = *(const uint4*)(&sP[lh * 40 + kb2 + 8]);
    }
    __syncthreads();

    // ---------- O += P V : 32 keys, 4 d-tiles, vT gives 2xb128 ----------
    BFrag fv[4];
#pragma unroll
    for (int t = 0; t < 4; ++t) {
      const int d = t * 16 + lh;
      const u16* vp = vT + ((size_t)(b * NHEAD + h) * DHEAD + d) * KLEN + j0 + hi;
      fv[t].q[0] = *(const uint4*)(vp);
      fv[t].q[1] = *(const uint4*)(vp + 8);
    }
#pragma unroll
    for (int t = 0; t < 4; ++t)
      o[t].v = __builtin_amdgcn_wmma_f32_16x16x32_bf16(
          false, fp.v, false, fv[t].v, (short)0, o[t].v, false, false);
  }

  // ---------- finalize: l = rowsum, write O/l as bf16 ----------
#pragma unroll
  for (int g = 0; g < 8; ++g) {
    float l = lrow[g];
    l += __shfl_xor(l, 1, 32);
    l += __shfl_xor(l, 2, 32);
    l += __shfl_xor(l, 4, 32);
    l += __shfl_xor(l, 8, 32);
    const float inv = 1.0f / l;
    const int row = i0 + g + (hi ? 8 : 0);
#pragma unroll
    for (int t = 0; t < 4; ++t)
      av[((size_t)row * BSZ + b) * HDH + h * DHEAD + t * 16 + lh] = f2bf(o[t].f[g] * inv);
  }
}

// =======================================================================
// out = LayerNorm(res + x) * g + b ; optional bf16 twin output for GEMM A.
// =======================================================================
__global__ __launch_bounds__(256) void ln_residual_kernel(
    const float* __restrict__ res, const float* __restrict__ x,
    const float* __restrict__ gw,  const float* __restrict__ bw,
    float* __restrict__ out, u16* __restrict__ outb)
{
  __shared__ float ssum[256], ssq[256];
  const int row = blockIdx.x;
  const int t   = threadIdx.x;
  float vals[4], s = 0.f, sq = 0.f;
#pragma unroll
  for (int k = 0; k < 4; ++k) {
    const int idx = t + k * 256;
    const float vv = res[(size_t)row * DMODEL + idx] + x[(size_t)row * DMODEL + idx];
    vals[k] = vv; s += vv; sq += vv * vv;
  }
  ssum[t] = s; ssq[t] = sq;
  __syncthreads();
  for (int off = 128; off > 0; off >>= 1) {
    if (t < off) { ssum[t] += ssum[t + off]; ssq[t] += ssq[t + off]; }
    __syncthreads();
  }
  const float mu  = ssum[0] * (1.f / DMODEL);
  const float var = ssq[0] * (1.f / DMODEL) - mu * mu;
  const float rs  = rsqrtf(var + 1e-5f);
#pragma unroll
  for (int k = 0; k < 4; ++k) {
    const int idx = t + k * 256;
    const float y = (vals[k] - mu) * rs * gw[idx] + bw[idx];
    out[(size_t)row * DMODEL + idx] = y;
    if (outb) outb[(size_t)row * DMODEL + idx] = f2bf(y);
  }
}

// =======================================================================
// Relative positional embedding, emitted directly as bf16 (GEMM A input).
// =======================================================================
__global__ void posemb_kernel(u16* __restrict__ r) {
  const int idx = blockIdx.x * blockDim.x + threadIdx.x;   // KLEN * 512
  const int m = idx >> 9;
  const int t = idx & 511;
  const float pos  = (float)(KLEN - 1 - m);
  const float invf = powf(10000.f, -((float)(2 * t)) / (float)DMODEL);
  const float ang  = pos * invf;
  r[(size_t)m * DMODEL + t]       = f2bf(sinf(ang));
  r[(size_t)m * DMODEL + 512 + t] = f2bf(cosf(ang));
}

// =======================================================================
extern "C" void kernel_launch(void* const* d_in, const int* in_sizes, int n_in,
                              void* d_out, int out_size, void* d_ws, size_t ws_size,
                              hipStream_t stream) {
  const float* x     = (const float*)d_in[0];
  const float* mems  = (const float*)d_in[1];
  const float* u     = (const float*)d_in[2];
  const float* v     = (const float*)d_in[3];
  const float* W_qkv = (const float*)d_in[4];
  const float* W_o   = (const float*)d_in[5];
  const float* W_r   = (const float*)d_in[6];
  const float* ln1_g = (const float*)d_in[7];
  const float* ln1_b = (const float*)d_in[8];
  const float* W1    = (const float*)d_in[9];
  const float* b1    = (const float*)d_in[10];
  const float* W2    = (const float*)d_in[11];
  const float* b2    = (const float*)d_in[12];
  const float* ln2_g = (const float*)d_in[13];
  const float* ln2_b = (const float*)d_in[14];
  float* out = (float*)d_out;

  // ---- workspace carve (256B aligned) ----
  char* base = (char*)d_ws;
  size_t off = 0;
  auto alloc = [&](size_t bytes) -> void* {
    void* p = base + off;
    off = (off + bytes + 255) & ~(size_t)255;
    return p;
  };
  u16*   r_bf     = (u16*)  alloc((size_t)KLEN * DMODEL * 2);
  u16*   wT       = (u16*)  alloc((size_t)DINNER * DMODEL * 2);       // max W^T
  u16*   heads_bf = (u16*)  alloc((size_t)KLEN * BSZ * QKV3 * 2);
  u16*   vT       = (u16*)  alloc((size_t)BSZ * HDH * KLEN * 2);
  u16*   rk_bf    = (u16*)  alloc((size_t)KLEN * HDH * 2);
  u16*   av_bf    = (u16*)  alloc((size_t)QLEN * BSZ * HDH * 2);
  float* tmp      = (float*)alloc((size_t)QLEN * BSZ * DMODEL * 4);
  float* hA       = (float*)alloc((size_t)QLEN * BSZ * DMODEL * 4);
  u16*   hA_bf    = (u16*)  alloc((size_t)QLEN * BSZ * DMODEL * 2);
  float* hB       = (float*)alloc((size_t)QLEN * BSZ * DMODEL * 4);
  u16*   hB_bf    = (u16*)  alloc((size_t)QLEN * BSZ * DMODEL * 2);
  u16*   ff1_bf   = (u16*)  alloc((size_t)QLEN * BSZ * DINNER * 2);
  u16*   x_bf     = (u16*)  alloc((size_t)QLEN * BSZ * DMODEL * 2);
  u16*   mems_bf  = (u16*)  alloc((size_t)NLAYER * MEMLEN * BSZ * DMODEL * 2);

  const int MQB = QLEN * BSZ;    // 2048 rows (i*B + b)
  const int MKB = KLEN * BSZ;    // 4096 rows (j*B + b)

  // ---- one-time preprocessing ----
  posemb_kernel<<<dim3(KLEN * 512 / 256), 256, 0, stream>>>(r_bf);
  {
    const int nx = QLEN * BSZ * DMODEL / 8;
    conv_bf16_kernel<<<(nx + 255) / 256, 256, 0, stream>>>(x, x_bf, nx);
    const int nm = NLAYER * MEMLEN * BSZ * DMODEL / 8;
    conv_bf16_kernel<<<(nm + 255) / 256, 256, 0, stream>>>(mems, mems_bf, nm);
  }

  const dim3 trBlk(32, 8);

  for (int i = 0; i < NLAYER; ++i) {
    const float* cur    = (i == 0) ? x    : hB;
    const u16*   cur_bf = (i == 0) ? x_bf : hB_bf;

    // heads = [mems_i || h] @ W_qkv[i]  (bf16 out)
    convtr_kernel<<<dim3(QKV3 / 32, DMODEL / 32), trBlk, 0, stream>>>(
        W_qkv + (size_t)i * DMODEL * QKV3, wT, DMODEL, QKV3);
    gemm_bf16_kernel<<<dim3(QKV3 / 64, MKB / 32), 32, 0, stream>>>(
        mems_bf + (size_t)i * MEMLEN * BSZ * DMODEL, MEMLEN * BSZ, cur_bf,
        wT, nullptr, nullptr, heads_bf, QKV3, DMODEL, 0);

    // rk = r @ W_r[i]  (bf16 out)
    convtr_kernel<<<dim3(HDH / 32, DMODEL / 32), trBlk, 0, stream>>>(
        W_r + (size_t)i * DMODEL * HDH, wT, DMODEL, HDH);
    gemm_bf16_kernel<<<dim3(HDH / 64, KLEN / 32), 32, 0, stream>>>(
        r_bf, KLEN, r_bf, wT, nullptr, nullptr, rk_bf, HDH, DMODEL, 0);

    // v relayout for P@V, then fused rel-attention
    vtrans_kernel<<<dim3(KLEN / 64, NHEAD, BSZ), 256, 0, stream>>>(heads_bf, vT);
    flash_attn_kernel<<<dim3(QLEN / 16, NHEAD, BSZ), 32, 0, stream>>>(
        heads_bf, rk_bf, vT, u, v, av_bf);

    // tmp = av @ W_o[i]  (f32 out for residual)
    convtr_kernel<<<dim3(DMODEL / 32, HDH / 32), trBlk, 0, stream>>>(
        W_o + (size_t)i * HDH * DMODEL, wT, HDH, DMODEL);
    gemm_bf16_kernel<<<dim3(DMODEL / 64, MQB / 32), 32, 0, stream>>>(
        av_bf, MQB, av_bf, wT, nullptr, tmp, nullptr, DMODEL, HDH, 0);

    // hA = LN(cur + tmp)  (f32 + bf16 twin)
    ln_residual_kernel<<<MQB, 256, 0, stream>>>(
        cur, tmp, ln1_g + i * DMODEL, ln1_b + i * DMODEL, hA, hA_bf);

    // ff1 = relu(hA @ W1[i] + b1[i])  (bf16 out)
    convtr_kernel<<<dim3(DINNER / 32, DMODEL / 32), trBlk, 0, stream>>>(
        W1 + (size_t)i * DMODEL * DINNER, wT, DMODEL, DINNER);
    gemm_bf16_kernel<<<dim3(DINNER / 64, MQB / 32), 32, 0, stream>>>(
        hA_bf, MQB, hA_bf, wT, b1 + i * DINNER, nullptr, ff1_bf, DINNER, DMODEL, 1);

    // tmp = ff1 @ W2[i] + b2[i]  (f32 out)
    convtr_kernel<<<dim3(DMODEL / 32, DINNER / 32), trBlk, 0, stream>>>(
        W2 + (size_t)i * DINNER * DMODEL, wT, DINNER, DMODEL);
    gemm_bf16_kernel<<<dim3(DMODEL / 64, MQB / 32), 32, 0, stream>>>(
        ff1_bf, MQB, ff1_bf, wT, b2 + i * DMODEL, tmp, nullptr, DMODEL, DINNER, 0);

    // next h = LN(hA + tmp); last layer -> d_out (no bf16 twin needed)
    float* nxt    = (i == NLAYER - 1) ? out : hB;
    u16*   nxt_bf = (i == NLAYER - 1) ? (u16*)nullptr : hB_bf;
    ln_residual_kernel<<<MQB, 256, 0, stream>>>(
        hA, tmp, ln2_g + i * DMODEL, ln2_b + i * DMODEL, nxt, nxt_bf);
  }
}
